// GlobalAttentionDecoder_22840636080331
// MI455X (gfx1250) — matrix operand order repaired and checked by reading, not compiled, not run on hardware
//
#include <hip/hip_runtime.h>
#include <hip/hip_bf16.h>

// ---------------------------------------------------------------------------
// GlobalAttentionDecoder on MI455X (gfx1250, wave32, WMMA)
//   V=32000, H=1000 (padded to 1024), N=64, L=50, T=51 steps, 2 LSTM layers.
//   All weight matrices converted once to bf16 with zero K-padding -> ~107 MB,
//   resident in the 192 MB L2 across the 51 sequential steps.
//   All matmuls via V_WMMA_F32_16X16X32_BF16; each wave computes a 16x(16*NT)
//   strip so the A fragment is loaded once and reused NT times per k-step.
// ---------------------------------------------------------------------------

typedef __attribute__((ext_vector_type(16))) __bf16       v16bf;
typedef __attribute__((ext_vector_type(8)))  float        v8f;
typedef __attribute__((ext_vector_type(4)))  unsigned int v4u;

union Frag32B { v4u u[2]; v16bf v; };

#define MODE_BIAS  1
#define MODE_TANH  2
#define MODE_WF32  4
#define MODE_WBF16 8

#define HH   1000
#define HP   1024
#define NB   64
#define LL   50
#define TT   51
#define VV   32000
#define G4   4000   // 4*H, multiple of 16

__device__ __forceinline__ float sigmoidf_(float x) { return 1.f / (1.f + __expf(-x)); }

// Inner K-loop: A fragment loaded once per k-step, reused against NT B tiles.
// A-fragment layout (ISA 7.12.2, 16-bit A 16x32): lane(0..15)=row, holds K
//   [kb + half*8 .. +7] in v0..3 and [kb+16+half*8 .. +7] in v4..7.
// B-fragment: lane&15 = output column, same K order as A (half*8.., 16+half*8..).
template <int NT>
__device__ __forceinline__ void wmma_k_loop(const __bf16* __restrict__ arow,
                                            const __bf16* __restrict__ bbase,
                                            long long ldb, int K, int half, int r16,
                                            v8f* acc) {
    for (int k = 0; k < K; k += 32) {
        Frag32B a;
        a.u[0] = *reinterpret_cast<const v4u*>(arow + k + half * 8);
        a.u[1] = *reinterpret_cast<const v4u*>(arow + k + 16 + half * 8);
#pragma unroll
        for (int j = 0; j < NT; ++j) {
            const __bf16* brow = bbase + (long long)(j * 16 + r16) * ldb;
            __builtin_prefetch((const void*)(brow + k + 32), 0, 1);  // speculative
            Frag32B b;
            b.u[0] = *reinterpret_cast<const v4u*>(brow + k + half * 8);
            b.u[1] = *reinterpret_cast<const v4u*>(brow + k + 16 + half * 8);
            acc[j] = __builtin_amdgcn_wmma_f32_16x16x32_bf16(
                /*neg_a=*/false, a.v, /*neg_b=*/false, b.v,
                /*c_mod=*/(short)0, acc[j], /*reuse_a=*/false, /*reuse_b=*/false);
        }
    }
}

// C(64 x Nout) = A(64 x K) @ B(Nout x K)^T  [+ A2 @ B2^T] [+ bias1+bias2]
// [tanh] -> optional f32 C and/or bf16 Cbf.  One 16x(16*NT) strip per wave;
// 4 waves per block cover M=64; blockIdx.x = strip index (Nout/(16*NT) blocks).
template <int NT>
__global__ void __launch_bounds__(128) wmma_gemm_bf16(
    const __bf16* __restrict__ A,  const __bf16* __restrict__ B,  int lda,  int ldb,  int K,
    const __bf16* __restrict__ A2, const __bf16* __restrict__ B2, int lda2, int ldb2, int K2,
    const float* __restrict__ bias1, const float* __restrict__ bias2,
    float* __restrict__ C, long long ldc,
    __bf16* __restrict__ Cbf, int ldcbf, int mode)
{
    const int wave = threadIdx.x >> 5;          // row tile 0..3 (M=64)
    const int lane = threadIdx.x & 31;
    const int r16  = lane & 15;
    const int half = lane >> 4;
    const int nb   = blockIdx.x * (16 * NT);    // output column base of strip

    v8f acc[NT];
#pragma unroll
    for (int j = 0; j < NT; ++j) acc[j] = (v8f){};

    wmma_k_loop<NT>(A + (long long)(wave * 16 + r16) * lda,
                    B + (long long)nb * ldb, ldb, K, half, r16, acc);
    if (A2) {
        wmma_k_loop<NT>(A2 + (long long)(wave * 16 + r16) * lda2,
                        B2 + (long long)nb * ldb2, ldb2, K2, half, r16, acc);
    }

#pragma unroll
    for (int j = 0; j < NT; ++j) {
        const int col = nb + j * 16 + r16;
        float badd = 0.f;
        if (mode & MODE_BIAS) badd = bias1[col] + bias2[col];
#pragma unroll
        for (int r = 0; r < 8; ++r) {           // C layout: VGPR r -> row r / r+8
            const int m = wave * 16 + r + half * 8;
            float v = acc[j][r] + badd;
            if (mode & MODE_TANH)  v = tanhf(v);
            if (mode & MODE_WF32)  C[(long long)m * ldc + col] = v;
            if (mode & MODE_WBF16) Cbf[m * ldcbf + col] = (__bf16)v;
        }
    }
}

// Zero-padded f32 -> bf16 weight conversion; supports 2 K-segments so the
// [emb | attn] / [ctx | out] concat halves each land on a 1024-aligned slot.
__global__ void convert_weight_kernel(const float* __restrict__ src, __bf16* __restrict__ dst,
                                      int Nsrc, int Npad, int segSrc, int segPad, int nseg)
{
    long long idx = (long long)blockIdx.x * 256 + threadIdx.x;
    const long long total = (long long)Npad * segPad * nseg;
    if (idx >= total) return;
    const int kp = segPad * nseg;
    const int n  = (int)(idx / kp);
    const int kc = (int)(idx % kp);
    const int s  = kc / segPad;
    const int k  = kc % segPad;
    float v = 0.f;
    if (n < Nsrc && k < segSrc)
        v = src[(long long)n * segSrc * nseg + (long long)s * segSrc + k];
    dst[idx] = (__bf16)v;
}

__global__ void init_state_kernel(const float* __restrict__ h_0, const float* __restrict__ c_0,
                                  float* h0s, float* c0s, float* h1s, float* c1s,
                                  __bf16* attnb, __bf16* catb)
{
    int idx = blockIdx.x * 256 + threadIdx.x;   // 64*1024
    if (idx >= NB * HP) return;
    const int n = idx >> 10, c = idx & (HP - 1);
    const bool in = (c < HH);
    h0s[idx] = in ? h_0[n * HH + c]            : 0.f;
    h1s[idx] = in ? h_0[NB * HH + n * HH + c]  : 0.f;
    c0s[idx] = in ? c_0[n * HH + c]            : 0.f;
    c1s[idx] = in ? c_0[NB * HH + n * HH + c]  : 0.f;
    attnb[idx] = (__bf16)0.f;
    catb[n * 2048 + c]      = (__bf16)0.f;
    catb[n * 2048 + HP + c] = (__bf16)0.f;
}

// Per step: xb = bf16([E[tok] | attn_vec]) padded; snapshot h0,h1 as bf16.
__global__ void prep_step_kernel(const float* __restrict__ E, const int* __restrict__ target, int t,
                                 const float* __restrict__ h0s, const float* __restrict__ h1s,
                                 const __bf16* __restrict__ attnb,
                                 __bf16* __restrict__ xb, __bf16* __restrict__ h0b,
                                 __bf16* __restrict__ h1b)
{
    int idx = blockIdx.x * 256 + threadIdx.x;   // 64*2048
    if (idx >= NB * 2 * HP) return;
    const int n = idx >> 11, col = idx & (2 * HP - 1);
    if (col < HP) {
        const int tok = target[n * 52 + t];
        xb[idx] = (__bf16)((col < HH) ? E[(long long)tok * HH + col] : 0.f);
        h0b[n * HP + col] = (__bf16)h0s[n * HP + col];
        h1b[n * HP + col] = (__bf16)h1s[n * HP + col];
    } else {
        xb[idx] = attnb[n * HP + (col - HP)];
    }
}

// LSTM pointwise: torch gate order i,f,g,o at offsets 0,H,2H,3H of (64 x 4000).
__global__ void lstm_act_kernel(const float* __restrict__ gates,
                                float* __restrict__ hs, float* __restrict__ cs,
                                __bf16* __restrict__ hb)
{
    int idx = blockIdx.x * 256 + threadIdx.x;   // 64*1000
    if (idx >= NB * HH) return;
    const int n = idx / HH, c = idx % HH;
    const float* g = gates + (long long)n * G4;
    const float i = sigmoidf_(g[c]);
    const float f = sigmoidf_(g[HH + c]);
    const float gg = tanhf(g[2 * HH + c]);
    const float o = sigmoidf_(g[3 * HH + c]);
    const float cn = f * cs[n * HP + c] + i * gg;
    cs[n * HP + c] = cn;
    const float h = o * tanhf(cn);
    hs[n * HP + c] = h;
    if (hb) hb[n * HP + c] = (__bf16)h;
}

// Dot attention + softmax + context; emits catb = bf16([ctx | out]) padded.
__global__ void __launch_bounds__(256) attention_kernel(
    const float* __restrict__ enc, const float* __restrict__ h1s,
    __bf16* __restrict__ catb)
{
    __shared__ float s_out[HH];
    __shared__ float s_align[64];
    const int n = blockIdx.x;
    for (int h = threadIdx.x; h < HH; h += 256) s_out[h] = h1s[n * HP + h];
    __syncthreads();
    const int wave = threadIdx.x >> 5, lane = threadIdx.x & 31;
    for (int l = wave; l < LL; l += 8) {
        const float* e = enc + ((long long)n * LL + l) * HH;
        float s = 0.f;
        for (int h = lane; h < HH; h += 32) s += e[h] * s_out[h];
        for (int off = 16; off > 0; off >>= 1) s += __shfl_xor(s, off, 32);
        if (lane == 0) s_align[l] = s;
    }
    __syncthreads();
    if (threadIdx.x == 0) {
        float mx = -1e30f;
        for (int l = 0; l < LL; ++l) mx = fmaxf(mx, s_align[l]);
        float sum = 0.f;
        for (int l = 0; l < LL; ++l) { float e = __expf(s_align[l] - mx); s_align[l] = e; sum += e; }
        const float inv = 1.f / sum;
        for (int l = 0; l < LL; ++l) s_align[l] *= inv;
    }
    __syncthreads();
    for (int h = threadIdx.x; h < HH; h += 256) {
        float c = 0.f;
        for (int l = 0; l < LL; ++l)
            c += s_align[l] * enc[((long long)n * LL + l) * HH + h];
        catb[n * 2 * HP + h]      = (__bf16)c;         // ctx
        catb[n * 2 * HP + HP + h] = (__bf16)s_out[h];  // out
    }
}

__global__ void finalize_kernel(const float* h0s, const float* h1s,
                                const float* c0s, const float* c1s,
                                float* __restrict__ out)
{
    int idx = blockIdx.x * 256 + threadIdx.x;   // 64*1000
    if (idx >= NB * HH) return;
    const int n = idx / HH, c = idx % HH;
    const long long base = (long long)NB * TT * VV;
    out[base + idx]                 = h0s[n * HP + c];
    out[base + NB * HH + idx]       = h1s[n * HP + c];
    out[base + 2 * NB * HH + idx]   = c0s[n * HP + c];
    out[base + 3 * NB * HH + idx]   = c1s[n * HP + c];
}

// ---------------------------------------------------------------------------
extern "C" void kernel_launch(void* const* d_in, const int* in_sizes, int n_in,
                              void* d_out, int out_size, void* d_ws, size_t ws_size,
                              hipStream_t stream) {
    const float* enc   = (const float*)d_in[0];
    const float* h_0   = (const float*)d_in[1];
    const float* c_0   = (const float*)d_in[2];
    const int*   tgt   = (const int*)  d_in[3];
    const float* E     = (const float*)d_in[4];
    const float* W_ih0 = (const float*)d_in[5];
    const float* W_hh0 = (const float*)d_in[6];
    const float* b_ih0 = (const float*)d_in[7];
    const float* b_hh0 = (const float*)d_in[8];
    const float* W_ih1 = (const float*)d_in[9];
    const float* W_hh1 = (const float*)d_in[10];
    const float* b_ih1 = (const float*)d_in[11];
    const float* b_hh1 = (const float*)d_in[12];
    const float* W_c   = (const float*)d_in[13];
    const float* W_out = (const float*)d_in[14];
    float* out = (float*)d_out;

    // ---- workspace carve-up (256B aligned) ----
    char* ws = (char*)d_ws;
    size_t off = 0;
    auto alloc = [&](size_t bytes) -> char* {
        char* p = ws + off;
        off += (bytes + 255) & ~(size_t)255;
        return p;
    };
    __bf16* Wih0b = (__bf16*)alloc((size_t)G4 * 2048 * 2);
    __bf16* Whh0b = (__bf16*)alloc((size_t)G4 * HP * 2);
    __bf16* Wih1b = (__bf16*)alloc((size_t)G4 * HP * 2);
    __bf16* Whh1b = (__bf16*)alloc((size_t)G4 * HP * 2);
    __bf16* Wcb   = (__bf16*)alloc((size_t)HP * 2048 * 2);
    __bf16* Woutb = (__bf16*)alloc((size_t)VV * HP * 2);
    __bf16* xb    = (__bf16*)alloc((size_t)NB * 2048 * 2);
    __bf16* h0b   = (__bf16*)alloc((size_t)NB * HP * 2);
    __bf16* h1b   = (__bf16*)alloc((size_t)NB * HP * 2);
    __bf16* attnb = (__bf16*)alloc((size_t)NB * HP * 2);
    __bf16* catb  = (__bf16*)alloc((size_t)NB * 2048 * 2);
    float*  gates0= (float*) alloc((size_t)NB * G4 * 4);
    float*  gates1= (float*) alloc((size_t)NB * G4 * 4);
    float*  h0s   = (float*) alloc((size_t)NB * HP * 4);
    float*  c0s   = (float*) alloc((size_t)NB * HP * 4);
    float*  h1s   = (float*) alloc((size_t)NB * HP * 4);
    float*  c1s   = (float*) alloc((size_t)NB * HP * 4);
    if (off > ws_size) return;  // workspace too small; bail deterministically

    auto cvt = [&](const float* s, __bf16* d, int Nsrc, int Npad, int nseg) {
        long long total = (long long)Npad * HP * nseg;
        convert_weight_kernel<<<(unsigned)((total + 255) / 256), 256, 0, stream>>>(
            s, d, Nsrc, Npad, HH, HP, nseg);
    };
    cvt(W_ih0, Wih0b, G4, G4, 2);     // (4000 x 2000) -> (4000 x 2048)
    cvt(W_hh0, Whh0b, G4, G4, 1);     // (4000 x 1000) -> (4000 x 1024)
    cvt(W_ih1, Wih1b, G4, G4, 1);
    cvt(W_hh1, Whh1b, G4, G4, 1);
    cvt(W_c,   Wcb,   HH, HP, 2);     // (1000 x 2000) -> (1024 x 2048)
    cvt(W_out, Woutb, VV, VV, 1);     // (32000 x 1000) -> (32000 x 1024)

    init_state_kernel<<<(NB * HP + 255) / 256, 256, 0, stream>>>(
        h_0, c_0, h0s, c0s, h1s, c1s, attnb, catb);

    for (int t = 0; t < TT; ++t) {
        prep_step_kernel<<<(NB * 2 * HP + 255) / 256, 256, 0, stream>>>(
            E, tgt, t, h0s, h1s, attnb, xb, h0b, h1b);

        // gates0 = xb @ Wih0^T + h0 @ Whh0^T + b_ih0 + b_hh0   (64 x 4000)
        wmma_gemm_bf16<2><<<G4 / 32, 128, 0, stream>>>(
            xb, Wih0b, 2048, 2048, 2048,
            h0b, Whh0b, HP, HP, HP,
            b_ih0, b_hh0, gates0, G4, nullptr, 0, MODE_BIAS | MODE_WF32);

        lstm_act_kernel<<<(NB * HH + 255) / 256, 256, 0, stream>>>(
            gates0, h0s, c0s, h0b);   // h0b <- bf16(new h0) for layer-1 GEMM

        // gates1 = h0 @ Wih1^T + h1 @ Whh1^T + b_ih1 + b_hh1   (64 x 4000)
        wmma_gemm_bf16<2><<<G4 / 32, 128, 0, stream>>>(
            h0b, Wih1b, HP, HP, HP,
            h1b, Whh1b, HP, HP, HP,
            b_ih1, b_hh1, gates1, G4, nullptr, 0, MODE_BIAS | MODE_WF32);

        lstm_act_kernel<<<(NB * HH + 255) / 256, 256, 0, stream>>>(
            gates1, h1s, c1s, nullptr);

        attention_kernel<<<NB, 256, 0, stream>>>(enc, h1s, catb);

        // attn = tanh([ctx|out] @ W_c^T) -> bf16 attnb   (64 x 1024)
        wmma_gemm_bf16<4><<<HP / 64, 128, 0, stream>>>(
            catb, Wcb, 2048, 2048, 2048,
            nullptr, nullptr, 0, 0, 0,
            nullptr, nullptr, nullptr, 0, attnb, HP, MODE_TANH | MODE_WBF16);

        // logits = attn @ W_out^T -> d_out[:, t, :]   (64 x 32000)
        wmma_gemm_bf16<4><<<VV / 64, 128, 0, stream>>>(
            attnb, Woutb, HP, HP, HP,
            nullptr, nullptr, 0, 0, 0,
            nullptr, nullptr,
            out + (long long)t * VV, (long long)TT * VV,
            nullptr, 0, MODE_WF32);
    }

    finalize_kernel<<<(NB * HH + 255) / 256, 256, 0, stream>>>(h0s, h1s, c0s, c1s, out);
}
